// LSTMModel_40578851012842
// MI455X (gfx1250) — matrix operand order repaired and checked by reading, not compiled, not run on hardware
//
#include <hip/hip_runtime.h>
#include <hip/hip_bf16.h>
#include <stdint.h>
#include <stddef.h>

// ---------------------------------------------------------------------------
// 2-layer LSTM (B=1024, T=512, I=64, H=256) + FC(256->1), CDNA5 WMMA f16/f32.
// ---------------------------------------------------------------------------

typedef __attribute__((ext_vector_type(16))) _Float16 v16h;
typedef __attribute__((ext_vector_type(8)))  _Float16 v8h;
typedef __attribute__((ext_vector_type(8)))  float    v8f;

#define BSZ 1024
#define TT  512
#define IN  64
#define HD  256
#define G4H 1024   // 4*H

union HF { v16h v; v8h h[2]; };

// A-fragment (16x32 f16, row-major source): lane group lo=lane&15 picks row,
// hi=lane>>4 picks the K sub-range: lanes 0-15 hold K=[k..k+16),
// lanes 16-31 hold K=[k+16..k+32), contiguous (matches B).
// Caller passes k0 = kb + hi*16; chunks at k0 and k0+8.
__device__ __forceinline__ v16h load_a_frag(const _Float16* __restrict__ row, int k0) {
  HF u;
  u.h[0] = *(const v8h*)(row + k0);
  u.h[1] = *(const v8h*)(row + k0 + 8);
  return u.v;
}

// B-fragment (32x16 f16): lane lo = column n; lanes 0-15 hold K=[kb..kb+16),
// lanes 16-31 hold K=[kb+16..kb+32), contiguous. Since gates = A @ W^T and W is
// stored [N,K] row-major, W's row n IS column n of B: 16 contiguous halves.
__device__ __forceinline__ v16h load_b_frag(const _Float16* __restrict__ p) {
  HF u;
  u.h[0] = *(const v8h*)(p);
  u.h[1] = *(const v8h*)(p + 8);
  return u.v;
}

__device__ __forceinline__ float sigmoidf_(float x) {
  return 1.0f / (1.0f + __expf(-x));
}

// One LSTM timestep. gates[B,4H] = x_t @ w_ih^T + h @ w_hh^T + bias, then
// pointwise cell update. Each 32-thread wave computes a 16x16 tile of h for
// all 4 gates. grid = (B/16, H/16), block = 32.
__global__ __launch_bounds__(32)
void lstm_step_kernel(const _Float16* __restrict__ xbase, int xstride, int Kx,
                      const _Float16* __restrict__ wih,   // [4H, Kx] row-major f16
                      const _Float16* __restrict__ whh,   // [4H, H]  row-major f16
                      const float*    __restrict__ bias,  // [4H] = b_ih + b_hh
                      const _Float16* __restrict__ hin,   // [B, H] f16
                      _Float16*       __restrict__ hout,  // [B, H] f16
                      float*          __restrict__ cstate,// [B, H] f32 (in/out)
                      float*          __restrict__ hout32)// optional f32 mirror
{
  const int lane = threadIdx.x;
  const int lo   = lane & 15;
  const int hi   = lane >> 4;
  const int mr   = blockIdx.x << 4;   // batch-row tile base
  const int jc   = blockIdx.y << 4;   // h-column tile base

  v8f acc0 = {}, acc1 = {}, acc2 = {}, acc3 = {};  // i, f, g, o

  // ---- input projection: x_t @ w_ih^T ----
  {
    const _Float16* arow = xbase + (size_t)(mr + lo) * xstride;
    const _Float16* w0 = wih + (size_t)(0 * HD + jc + lo) * Kx;
    const _Float16* w1 = wih + (size_t)(1 * HD + jc + lo) * Kx;
    const _Float16* w2 = wih + (size_t)(2 * HD + jc + lo) * Kx;
    const _Float16* w3 = wih + (size_t)(3 * HD + jc + lo) * Kx;
    for (int kb = 0; kb < Kx; kb += 32) {
      v16h a = load_a_frag(arow, kb + hi * 16);
      const int ko = kb + hi * 16;
      acc0 = __builtin_amdgcn_wmma_f32_16x16x32_f16(false, a, false, load_b_frag(w0 + ko), (short)0, acc0, false, false);
      acc1 = __builtin_amdgcn_wmma_f32_16x16x32_f16(false, a, false, load_b_frag(w1 + ko), (short)0, acc1, false, false);
      acc2 = __builtin_amdgcn_wmma_f32_16x16x32_f16(false, a, false, load_b_frag(w2 + ko), (short)0, acc2, false, false);
      acc3 = __builtin_amdgcn_wmma_f32_16x16x32_f16(false, a, false, load_b_frag(w3 + ko), (short)0, acc3, false, false);
    }
  }
  // ---- recurrent projection: h @ w_hh^T ----
  {
    const _Float16* arow = hin + (size_t)(mr + lo) * HD;
    const _Float16* u0 = whh + (size_t)(0 * HD + jc + lo) * HD;
    const _Float16* u1 = whh + (size_t)(1 * HD + jc + lo) * HD;
    const _Float16* u2 = whh + (size_t)(2 * HD + jc + lo) * HD;
    const _Float16* u3 = whh + (size_t)(3 * HD + jc + lo) * HD;
    for (int kb = 0; kb < HD; kb += 32) {
      v16h a = load_a_frag(arow, kb + hi * 16);
      const int ko = kb + hi * 16;
      acc0 = __builtin_amdgcn_wmma_f32_16x16x32_f16(false, a, false, load_b_frag(u0 + ko), (short)0, acc0, false, false);
      acc1 = __builtin_amdgcn_wmma_f32_16x16x32_f16(false, a, false, load_b_frag(u1 + ko), (short)0, acc1, false, false);
      acc2 = __builtin_amdgcn_wmma_f32_16x16x32_f16(false, a, false, load_b_frag(u2 + ko), (short)0, acc2, false, false);
      acc3 = __builtin_amdgcn_wmma_f32_16x16x32_f16(false, a, false, load_b_frag(u3 + ko), (short)0, acc3, false, false);
    }
  }

  // C/D layout: acc[j] at this lane = element (m = j + 8*hi, n = lo). All four
  // gate values for a given (m,n) live in the same lane & index -> pure
  // register-resident pointwise cell update.
  const int col = jc + lo;
  const float bi = bias[0 * HD + col];
  const float bf = bias[1 * HD + col];
  const float bg = bias[2 * HD + col];
  const float bo = bias[3 * HD + col];
#pragma unroll
  for (int j = 0; j < 8; ++j) {
    const int brow = mr + j + (hi << 3);
    const size_t idx = (size_t)brow * HD + col;
    const float iv = sigmoidf_(acc0[j] + bi);
    const float fv = sigmoidf_(acc1[j] + bf);
    const float gv = tanhf(acc2[j] + bg);
    const float ov = sigmoidf_(acc3[j] + bo);
    const float cn = fv * cstate[idx] + iv * gv;
    cstate[idx] = cn;
    const float hn = ov * tanhf(cn);
    hout[idx] = (_Float16)hn;
    if (hout32) hout32[idx] = hn;
  }
}

// ---------------------------------------------------------------------------
// Helper kernels
// ---------------------------------------------------------------------------

__global__ void zero_kernel(uint32_t* __restrict__ p, int n4) {
  int i = blockIdx.x * blockDim.x + threadIdx.x;
  if (i < n4) p[i] = 0u;
}

__global__ void f32_to_f16_kernel(const float* __restrict__ src,
                                  _Float16* __restrict__ dst, int n) {
  int i = (blockIdx.x * blockDim.x + threadIdx.x) * 4;
  if (i < n) {
    float4 v = *(const float4*)(src + i);
    dst[i + 0] = (_Float16)v.x;
    dst[i + 1] = (_Float16)v.y;
    dst[i + 2] = (_Float16)v.z;
    dst[i + 3] = (_Float16)v.w;
  }
}

__global__ void bias_add_kernel(const float* __restrict__ a,
                                const float* __restrict__ b,
                                float* __restrict__ o, int n) {
  int i = blockIdx.x * blockDim.x + threadIdx.x;
  if (i < n) o[i] = a[i] + b[i];
}

// out[b] = dot(h[b,:], w) + bias ; one wave per batch row (wave32).
__global__ void fc_kernel(const float* __restrict__ h,
                          const float* __restrict__ w,
                          const float* __restrict__ bias,
                          float* __restrict__ out) {
  const int wave = threadIdx.x >> 5;
  const int lane = threadIdx.x & 31;
  const int b = blockIdx.x * (blockDim.x >> 5) + wave;
  float s = 0.0f;
  for (int k = lane; k < HD; k += 32) s += h[(size_t)b * HD + k] * w[k];
  for (int off = 16; off > 0; off >>= 1) s += __shfl_down(s, off, 32);
  if (lane == 0) out[b] = s + bias[0];
}

// ---------------------------------------------------------------------------
// Launch
// ---------------------------------------------------------------------------

static inline size_t align_up(size_t x, size_t a) { return (x + a - 1) & ~(a - 1); }

extern "C" void kernel_launch(void* const* d_in, const int* in_sizes, int n_in,
                              void* d_out, int out_size, void* d_ws, size_t ws_size,
                              hipStream_t stream) {
  (void)in_sizes; (void)n_in; (void)out_size; (void)ws_size;

  const float* x     = (const float*)d_in[0];   // [B,T,I]
  const float* w_ih0 = (const float*)d_in[1];   // [4H,I]
  const float* w_hh0 = (const float*)d_in[2];   // [4H,H]
  const float* b_ih0 = (const float*)d_in[3];
  const float* b_hh0 = (const float*)d_in[4];
  const float* w_ih1 = (const float*)d_in[5];   // [4H,H]
  const float* w_hh1 = (const float*)d_in[6];   // [4H,H]
  const float* b_ih1 = (const float*)d_in[7];
  const float* b_hh1 = (const float*)d_in[8];
  const float* fc_w  = (const float*)d_in[9];   // [1,H]
  const float* fc_b  = (const float*)d_in[10];  // [1]
  float* out = (float*)d_out;                   // [B,1]

  // ---- workspace layout (256-byte aligned slabs) ----
  char* ws = (char*)d_ws;
  size_t off = 0;
  auto take = [&](size_t bytes) { size_t r = off; off += align_up(bytes, 256); return r; };

  // zero-at-start region (state buffers) — must be first & contiguous
  const size_t o_hzero = take((size_t)BSZ * HD * 2);   // layer0 h(-1), f16
  const size_t o_h1a   = take((size_t)BSZ * HD * 2);   // layer1 h ping, f16
  const size_t o_h1b   = take((size_t)BSZ * HD * 2);   // layer1 h pong, f16
  const size_t o_c0    = take((size_t)BSZ * HD * 4);   // layer0 c, f32
  const size_t o_c1    = take((size_t)BSZ * HD * 4);   // layer1 c, f32
  const size_t zero_bytes = off;

  const size_t o_xh    = take((size_t)BSZ * TT * IN * 2);  // x in f16
  const size_t o_wih0  = take((size_t)G4H * IN * 2);
  const size_t o_whh0  = take((size_t)G4H * HD * 2);
  const size_t o_wih1  = take((size_t)G4H * HD * 2);
  const size_t o_whh1  = take((size_t)G4H * HD * 2);
  const size_t o_bias0 = take((size_t)G4H * 4);
  const size_t o_bias1 = take((size_t)G4H * 4);
  const size_t o_h1f32 = take((size_t)BSZ * HD * 4);       // layer1 h, f32 mirror
  const size_t o_hseq0 = take((size_t)TT * BSZ * HD * 2);  // layer0 h sequence, f16

  _Float16* xh    = (_Float16*)(ws + o_xh);
  _Float16* wih0h = (_Float16*)(ws + o_wih0);
  _Float16* whh0h = (_Float16*)(ws + o_whh0);
  _Float16* wih1h = (_Float16*)(ws + o_wih1);
  _Float16* whh1h = (_Float16*)(ws + o_whh1);
  float*    bias0 = (float*)(ws + o_bias0);
  float*    bias1 = (float*)(ws + o_bias1);
  _Float16* hzero = (_Float16*)(ws + o_hzero);
  _Float16* h1a   = (_Float16*)(ws + o_h1a);
  _Float16* h1b   = (_Float16*)(ws + o_h1b);
  float*    c0    = (float*)(ws + o_c0);
  float*    c1    = (float*)(ws + o_c1);
  float*    h1f32 = (float*)(ws + o_h1f32);
  _Float16* hseq0 = (_Float16*)(ws + o_hseq0);

  // ---- prep: zero state, convert weights/x to f16, fuse biases ----
  {
    int n4 = (int)(zero_bytes / 4);
    zero_kernel<<<(n4 + 255) / 256, 256, 0, stream>>>((uint32_t*)ws, n4);
  }
  {
    int n = BSZ * TT * IN;
    f32_to_f16_kernel<<<(n / 4 + 255) / 256, 256, 0, stream>>>(x, xh, n);
  }
  {
    int n = G4H * IN;
    f32_to_f16_kernel<<<(n / 4 + 255) / 256, 256, 0, stream>>>(w_ih0, wih0h, n);
  }
  {
    int n = G4H * HD;
    f32_to_f16_kernel<<<(n / 4 + 255) / 256, 256, 0, stream>>>(w_hh0, whh0h, n);
    f32_to_f16_kernel<<<(n / 4 + 255) / 256, 256, 0, stream>>>(w_ih1, wih1h, n);
    f32_to_f16_kernel<<<(n / 4 + 255) / 256, 256, 0, stream>>>(w_hh1, whh1h, n);
  }
  bias_add_kernel<<<4, 256, 0, stream>>>(b_ih0, b_hh0, bias0, G4H);
  bias_add_kernel<<<4, 256, 0, stream>>>(b_ih1, b_hh1, bias1, G4H);

  const dim3 sgrid(BSZ / 16, HD / 16);

  // ---- layer 0: h chain lives directly in hseq0 slices ----
  for (int t = 0; t < TT; ++t) {
    const _Float16* hin = (t == 0) ? hzero : (hseq0 + (size_t)(t - 1) * BSZ * HD);
    _Float16* hout = hseq0 + (size_t)t * BSZ * HD;
    lstm_step_kernel<<<sgrid, 32, 0, stream>>>(
        xh + (size_t)t * IN, TT * IN, IN,
        wih0h, whh0h, bias0, hin, hout, c0, nullptr);
  }

  // ---- layer 1: input = hseq0[t], ping-pong h buffers, f32 mirror for FC ----
  for (int t = 0; t < TT; ++t) {
    const _Float16* hin = (t & 1) ? h1b : h1a;
    _Float16* hout      = (t & 1) ? h1a : h1b;
    lstm_step_kernel<<<sgrid, 32, 0, stream>>>(
        hseq0 + (size_t)t * BSZ * HD, HD, HD,
        wih1h, whh1h, bias1, hin, hout, c1, h1f32);
  }

  // ---- final FC on last hidden state ----
  fc_kernel<<<BSZ / 8, 256, 0, stream>>>(h1f32, fc_w, fc_b, out);
}
